// GraphAttentionLayer_88304527606281
// MI455X (gfx1250) — matrix operand order repaired and checked by reading, not compiled, not run on hardware
//
#include <hip/hip_runtime.h>
#include <hip/hip_bf16.h>

#define N_NODES 8192
#define FDIM    256
#define ALPHA   0.2f
#define MASK_VAL (-9.0e15f)
#define LOG2E   1.4426950408889634f

typedef __attribute__((ext_vector_type(16))) __bf16 v16bf;
typedef __attribute__((ext_vector_type(8)))  float  v8f;
typedef __attribute__((ext_vector_type(4)))  int    v4i;
typedef __attribute__((ext_vector_type(4)))  float  v4f;

// ---------------------------------------------------------------------------
// Kernel 1: h = x @ W (f32), store hT as bf16 (FDIM x N, K-major for WMMA B),
// and si = (h@a1)*log2(e), sj = (h@a2)*log2(e)  (base-2 softmax domain).
// One block = 16 rows of h, 256 threads = one feature column each.
// ---------------------------------------------------------------------------
__global__ __launch_bounds__(256)
void gat_hgemm(const float* __restrict__ x,
               const float* __restrict__ W,
               const float* __restrict__ a,
               __bf16* __restrict__ hT,
               float* __restrict__ si,
               float* __restrict__ sj)
{
    __shared__ float xs[16][FDIM];
    __shared__ float red1[256];
    __shared__ float red2[256];

    const int t  = threadIdx.x;       // feature column f
    const int i0 = blockIdx.x * 16;   // first row of this tile

    #pragma unroll
    for (int r = 0; r < 16; ++r)
        xs[r][t] = x[(size_t)(i0 + r) * FDIM + t];
    __syncthreads();

    float acc[16];
    #pragma unroll
    for (int r = 0; r < 16; ++r) acc[r] = 0.0f;

    for (int k = 0; k < FDIM; ++k) {
        const float w = W[(size_t)k * FDIM + t];   // coalesced row of W
        #pragma unroll
        for (int r = 0; r < 16; ++r)
            acc[r] = __builtin_fmaf(xs[r][k], w, acc[r]);
    }

    // Transposed bf16 copy: hT[f][i].  16 consecutive bf16 = one 32B store.
    v16bf pack;
    #pragma unroll
    for (int r = 0; r < 16; ++r) pack[r] = (__bf16)acc[r];
    *(v16bf*)(hT + (size_t)t * N_NODES + i0) = pack;

    // Attention projections si/sj (per-row dot over features), pre-scaled
    // into the exp2 domain.
    const float a1 = a[t];
    const float a2 = a[FDIM + t];
    for (int r = 0; r < 16; ++r) {
        red1[t] = acc[r] * a1;
        red2[t] = acc[r] * a2;
        __syncthreads();
        for (int s = 128; s > 0; s >>= 1) {
            if (t < s) { red1[t] += red1[t + s]; red2[t] += red2[t + s]; }
            __syncthreads();
        }
        if (t == 0) { si[i0 + r] = red1[0] * LOG2E; sj[i0 + r] = red2[0] * LOG2E; }
        __syncthreads();
    }
}

// ---------------------------------------------------------------------------
// Kernel 2: fused masked-softmax attention + h' = attn @ h + ELU.
// Workgroup = 4 waves, all owning the SAME 16 output rows; wave w covers
// feature blocks 4w..4w+3 (64 features -> 4 WMMA C tiles, 32 acc VGPRs).
// Softmax stats computed redundantly per wave (adj tile shared via WGP$).
// Online (flash) softmax in base-2 over j tiles of 32 (WMMA K=32, bf16).
// A layout (16-bit A 16x32): lane L = row m=L&15,
//   K = 16*hi+{0..15} in elems 0..15 (matches B).
// B (32x16): lane = column n, K = 16*hi+{0..15} -> one 32B load from hT.
// ---------------------------------------------------------------------------
__global__ __launch_bounds__(128)
void gat_attn(const int*    __restrict__ adj,
              const __bf16* __restrict__ hT,
              const float*  __restrict__ si,
              const float*  __restrict__ sj,
              float*        __restrict__ out)
{
    const int lane = threadIdx.x & 31;
    const int wave = threadIdx.x >> 5;   // 0..3: feature slice
    const int m    = lane & 15;
    const int hi   = lane >> 4;
    const int i0   = blockIdx.x * 16;

    const float si_r  = si[i0 + m];
    const int* adjRow = adj + (size_t)(i0 + m) * N_NODES;

    float Mrow = -INFINITY;   // running row max (lanes m and m+16 mirror row m)
    float Srow = 0.0f;        // running row sum of exp2

    v8f accb[4];
    #pragma unroll
    for (int b = 0; b < 4; ++b)
        #pragma unroll
        for (int r = 0; r < 8; ++r) accb[b][r] = 0.0f;

    for (int j0 = 0; j0 < N_NODES; j0 += 32) {
        const int k1 = j0 + 16 * hi;        // first 8 K of this lane
        const int k2 = j0 + 16 * hi + 8;    // second 8 K of this lane

        // adj is a 256MB read-once stream: keep it out of L2 (NT hint)
        const v4i av0 = __builtin_nontemporal_load((const v4i*)(adjRow + k1));
        const v4i av1 = __builtin_nontemporal_load((const v4i*)(adjRow + k1 + 4));
        const v4i av2 = __builtin_nontemporal_load((const v4i*)(adjRow + k2));
        const v4i av3 = __builtin_nontemporal_load((const v4i*)(adjRow + k2 + 4));
        const v4f sv0 = *(const v4f*)(sj + k1);
        const v4f sv1 = *(const v4f*)(sj + k1 + 4);
        const v4f sv2 = *(const v4f*)(sj + k2);
        const v4f sv3 = *(const v4f*)(sj + k2 + 4);

        int   ad[16];
        float sv[16];
        ad[0]=av0.x; ad[1]=av0.y; ad[2]=av0.z; ad[3]=av0.w;
        ad[4]=av1.x; ad[5]=av1.y; ad[6]=av1.z; ad[7]=av1.w;
        ad[8]=av2.x; ad[9]=av2.y; ad[10]=av2.z; ad[11]=av2.w;
        ad[12]=av3.x; ad[13]=av3.y; ad[14]=av3.z; ad[15]=av3.w;
        sv[0]=sv0.x; sv[1]=sv0.y; sv[2]=sv0.z; sv[3]=sv0.w;
        sv[4]=sv1.x; sv[5]=sv1.y; sv[6]=sv1.z; sv[7]=sv1.w;
        sv[8]=sv2.x; sv[9]=sv2.y; sv[10]=sv2.z; sv[11]=sv2.w;
        sv[12]=sv3.x; sv[13]=sv3.y; sv[14]=sv3.z; sv[15]=sv3.w;

        float e[16];
        float tmax = MASK_VAL;
        #pragma unroll
        for (int q = 0; q < 16; ++q) {
            float ee = si_r + sv[q];                  // already * log2(e)
            ee = (ee > 0.0f) ? ee : ALPHA * ee;       // LeakyReLU(0.2)
            ee = (ad[q] > 0) ? ee : MASK_VAL;         // adjacency mask
            e[q] = ee;
            tmax = fmaxf(tmax, ee);
        }
        // combine the two half-lanes holding the same row
        tmax = fmaxf(tmax, __shfl_xor(tmax, 16, 32));

        const float newM = fmaxf(Mrow, tmax);
        const bool  bump = newM > Mrow;
        if (__ballot(bump)) {
            // v_exp_f32 is natively 2^x; ==1 for unchanged rows
            const float scale = __builtin_amdgcn_exp2f(Mrow - newM);
            Srow *= scale;
            #pragma unroll
            for (int r = 0; r < 8; ++r) {
                // stats for C row (8*hi + r) live in lane (8*hi + r)
                const float sr = __shfl(scale, 8 * hi + r, 32);
                #pragma unroll
                for (int b = 0; b < 4; ++b) accb[b][r] *= sr;
            }
        }
        Mrow = newM;

        // probabilities -> WMMA A operand (bf16); exp2 == exp in orig domain
        v16bf A;
        float tsum = 0.0f;
        #pragma unroll
        for (int q = 0; q < 16; ++q) {
            const float p = __builtin_amdgcn_exp2f(e[q] - Mrow);
            tsum += p;
            A[q] = (__bf16)p;
        }
        Srow += tsum + __shfl_xor(tsum, 16, 32);

        // this wave's 4 feature blocks: C[16x16] += A[16x32] * hT-block[32x16]
        const __bf16* hTbase = hT + (size_t)(lane & 15) * N_NODES + j0 + 16 * hi
                                  + (size_t)(wave * 64) * N_NODES;
        #pragma unroll
        for (int b = 0; b < 4; ++b) {
            const v16bf B = *(const v16bf*)(hTbase + (size_t)(16 * b) * N_NODES);
            accb[b] = __builtin_amdgcn_wmma_f32_16x16x32_bf16(
                false, A, false, B, (short)0, accb[b], false, false);
        }
    }

    // epilogue: normalize by row sum, ELU, non-temporal store f32
    float Sr[8];
    #pragma unroll
    for (int r = 0; r < 8; ++r) Sr[r] = __shfl(Srow, 8 * hi + r, 32);

    const int n = lane & 15;
    #pragma unroll
    for (int b = 0; b < 4; ++b) {
        #pragma unroll
        for (int r = 0; r < 8; ++r) {
            const int row = i0 + 8 * hi + r;
            const int col = wave * 64 + 16 * b + n;
            float v = (Sr[r] > 0.0f) ? (accb[b][r] / Sr[r]) : 0.0f;
            v = (v > 0.0f) ? v : (__expf(v) - 1.0f);   // ELU
            __builtin_nontemporal_store(v, out + (size_t)row * FDIM + col);
        }
    }
}

// ---------------------------------------------------------------------------
extern "C" void kernel_launch(void* const* d_in, const int* in_sizes, int n_in,
                              void* d_out, int out_size, void* d_ws, size_t ws_size,
                              hipStream_t stream)
{
    const float* x   = (const float*)d_in[0];
    const int*   adj = (const int*)  d_in[1];
    const float* W   = (const float*)d_in[2];
    const float* a   = (const float*)d_in[3];
    float*       out = (float*)d_out;

    __bf16* hT = (__bf16*)d_ws;                                   // 4 MB
    float*  si = (float*)((char*)d_ws + (size_t)N_NODES * FDIM * sizeof(__bf16));
    float*  sj = si + N_NODES;                                    // +64 KB

    hipLaunchKernelGGL(gat_hgemm, dim3(N_NODES / 16), dim3(256), 0, stream,
                       x, W, a, hT, si, sj);
    hipLaunchKernelGGL(gat_attn, dim3(N_NODES / 16), dim3(128), 0, stream,
                       adj, hT, si, sj, out);
}
